// WaveRNN_22033182228938
// MI455X (gfx1250) — matrix operand order repaired and checked by reading, not compiled, not run on hardware
//
#include <hip/hip_runtime.h>
#include <cstdint>

// ---------------------------------------------------------------- constants
#define HID    896
#define HALF_  448
#define SEQ    2048
#define NB     16          // batch
#define BINS   256
#define NWG    56          // 896/16 persistent GRU workgroups
#define ROWS   (NB*SEQ)    // 32768

typedef __attribute__((ext_vector_type(16))) __bf16       v16bf;
typedef __attribute__((ext_vector_type(8)))  float        v8f;
typedef __attribute__((ext_vector_type(4)))  unsigned int v4u;
typedef __attribute__((ext_vector_type(8)))  int          v8i;
typedef __attribute__((ext_vector_type(4)))  int          v4i;

union FragBF { v4u q[2]; v16bf v; };

__device__ __forceinline__ unsigned short f2bf(float x) {
  unsigned u = __float_as_uint(x);
  u += 0x7fffu + ((u >> 16) & 1u);          // round-to-nearest-even
  return (unsigned short)(u >> 16);
}
// fast activations: v_exp_f32 + v_rcp_f32, no IEEE division fixup sequence
__device__ __forceinline__ float fsigm(float x) {
  return __builtin_amdgcn_rcpf(1.0f + __expf(-x));
}
__device__ __forceinline__ float ftanh(float x) {
  return 1.0f - 2.0f * __builtin_amdgcn_rcpf(1.0f + __expf(2.0f * x));
}

// A fragment (16x32 bf16, rows = M): lane<16 -> M=lane, K={k0..k0+7, k0+16..k0+23}
//                                    lane>=16 -> M=lane-16, K={k0+8..15, k0+24..31}
__device__ __forceinline__ v16bf loadA(const unsigned short* base, int lane, int k0, int ld) {
  FragBF f;
  int m  = lane & 15;
  int ko = k0 + ((lane & 16) ? 8 : 0);
  const unsigned short* p = base + m * ld + ko;
  f.q[0] = *reinterpret_cast<const v4u*>(p);
  f.q[1] = *reinterpret_cast<const v4u*>(p + 16);
  return f.v;
}
// B fragment (32x16 bf16, cols = N), storage is [n][k] (K contiguous per column):
// lane<16 -> N=lane, K={k0..k0+7, k0+16..k0+23} ; lane>=16 -> N=lane-16, K={k0+8..15, k0+24..31}
__device__ __forceinline__ v16bf loadB(const unsigned short* base, int lane, int k0, int ld) {
  FragBF f;
  int n  = lane & 15;
  int ko = k0 + ((lane & 16) ? 8 : 0);
  const unsigned short* p = base + n * ld + ko;
  f.q[0] = *reinterpret_cast<const v4u*>(p);
  f.q[1] = *reinterpret_cast<const v4u*>(p + 16);
  return f.v;
}

#if defined(__has_builtin)
#if __has_builtin(__builtin_amdgcn_tensor_load_to_lds)
#define HAVE_TDM 1
#endif
#endif

#ifdef HAVE_TDM
// One 2D TDM transfer: tile (tile_d0 x tile_d1) of a (dim0 x dim1) bf16 tensor -> LDS.
// Call from a single (uniform) wave; EXEC is ignored by TDM.
__device__ __forceinline__ void tdm_load_2d(unsigned lds_addr, const void* gptr,
                                            unsigned dim0, unsigned dim1,
                                            unsigned tile_d0, unsigned tile_d1,
                                            unsigned stride) {
  uint64_t ga = (uint64_t)(uintptr_t)gptr;
  v4u g0;
  g0[0] = 1u;                                              // count=1, user descriptor
  g0[1] = lds_addr;                                        // lds_addr (bytes)
  g0[2] = (unsigned)(ga & 0xffffffffu);                    // global_addr[31:0]
  g0[3] = (unsigned)((ga >> 32) & 0x1ffffffu) | (2u << 30);// global_addr[56:32] | type=2
  v8i g1;
  g1[0] = (int)(1u << 16);                                 // data_size=1 (2B), no flags
  g1[1] = (int)(dim0 << 16);                               // tensor_dim0[15:0]
  g1[2] = (int)((dim0 >> 16) & 0xffffu) | (int)(dim1 << 16);        // dim0 hi | dim1 lo
  g1[3] = (int)((dim1 >> 16) & 0xffffu) | (int)(tile_d0 << 16);     // dim1 hi | tile_dim0
  g1[4] = (int)(tile_d1 & 0xffffu);                        // tile_dim1, tile_dim2=0
  g1[5] = (int)stride;                                     // tensor_dim0_stride (elems)
  g1[6] = 0;
  g1[7] = 0;
  v4i z4 = {0, 0, 0, 0};
  v8i z8 = {0, 0, 0, 0, 0, 0, 0, 0};
  __builtin_amdgcn_tensor_load_to_lds(g0, g1, z4, z4, z8, 0);
}
#endif

// ---------------------------------------------------------------- prep kernels
__global__ void zero_init_kernel(unsigned short* h, int* ctr) {
  int i = blockIdx.x * blockDim.x + threadIdx.x;
  if (i < NB * HID) h[i] = 0;
  if (i == 0) ctr[0] = 0;
}
__global__ void conv_bf16_kernel(const float* __restrict__ src, unsigned short* __restrict__ dst, int n) {
  int i = blockIdx.x * blockDim.x + threadIdx.x;
  if (i < n) dst[i] = f2bf(src[i]);
}
// src is [K][N] row-major f32; dst is [N][K] bf16 (B-fragment friendly)
__global__ void transpose_bf16_kernel(const float* __restrict__ src, unsigned short* __restrict__ dst,
                                      int K, int N) {
  int i = blockIdx.x * blockDim.x + threadIdx.x;
  if (i < K * N) {
    int k = i / N, n = i % N;
    dst[n * K + k] = f2bf(src[i]);
  }
}

// ---------------------------------------------------------------- GRU persistent kernel
// grid = 56 WGs x 128 threads. WG jt owns hidden columns [16*jt, 16*jt+16).
// LDS: weights for its 3 gate tiles stay resident across all 2048 steps.
__global__ void gru_kernel(const float* __restrict__ cond,     // (16,2048,3,896)
                           const float* __restrict__ insig,    // (16,2048,2)
                           const float* __restrict__ tcoarse,  // (16,2048,1)
                           const float* __restrict__ W_c,      // (2,1344)
                           const float* __restrict__ W_f,      // (3,1344)
                           const float* __restrict__ b_ih,     // (2688)
                           const float* __restrict__ b_hh,     // (2688)
                           const unsigned short* __restrict__ whh_bf, // (2688,896) bf16
                           unsigned short* __restrict__ h_glob,       // (16,896) bf16
                           unsigned short* __restrict__ hc_out,       // (16,2048,448) bf16
                           int* __restrict__ ctr) {
  extern __shared__ unsigned char smem[];
  unsigned short* wB   = (unsigned short*)smem;                 // [3][16][896] bf16   86016 B
  unsigned short* hA   = wB + 3 * 16 * HID;                     // [16][896] bf16      28672 B
  float* cT    = (float*)(smem + 114688);                       // [3][256]
  float* xs    = cT + 768;                                      // [3][256]
  float* sigL  = xs + 768;                                      // [16][3]
  float* wsg   = sigL + 48;                                     // [3][3*16] signal weights
  float* bihL  = wsg + 144;                                     // [3][16]
  float* bhhL  = bihL + 48;                                     // [3][16]

  const int jt   = blockIdx.x;
  const int jc   = jt * 16;
  const int tid  = threadIdx.x;
  const int lane = tid & 31;
  const int wv   = tid >> 5;
  const bool fine = (jc >= HALF_);
  const int  cc   = jc - (fine ? HALF_ : 0);

  // ---- preload recurrent weight tiles into LDS (resident for the whole scan)
#ifdef HAVE_TDM
  const unsigned lds_wB = (unsigned)(uintptr_t)(void*)wB;
  const unsigned lds_hA = (unsigned)(uintptr_t)(void*)hA;
  if (wv == 0) {
    for (int g = 0; g < 3; ++g) {
      tdm_load_2d(lds_wB + (unsigned)(g * 16 * HID * 2),
                  whh_bf + (size_t)(g * HID + jc) * HID,
                  HID, 16, HID, 16, HID);
    }
    __builtin_amdgcn_s_wait_tensorcnt(0);
  }
#else
  {
    // fallback: plain vector copy (rows of each gate tile are contiguous)
    for (int g = 0; g < 3; ++g) {
      const v4u* s = (const v4u*)(whh_bf + (size_t)(g * HID + jc) * HID);
      v4u* d = (v4u*)(wB + g * 16 * HID);
      for (int r = 0; r < 14; ++r) d[tid + r * 128] = s[tid + r * 128];
    }
  }
#endif

  // ---- preload per-WG constants (signal-projection weight columns + biases)
  if (tid < 48) {
    int g = tid / 16, jj = tid % 16;
    bihL[tid] = b_ih[g * HID + jc + jj];
    bhhL[tid] = b_hh[g * HID + jc + jj];
    int col = g * HALF_ + cc + jj;           // column inside the (., 3*448) proj matrix
    if (!fine) {
      wsg[0 * 48 + tid] = W_c[col];
      wsg[1 * 48 + tid] = W_c[1344 + col];
      wsg[2 * 48 + tid] = 0.0f;
    } else {
      wsg[0 * 48 + tid] = W_f[col];
      wsg[1 * 48 + tid] = W_f[1344 + col];
      wsg[2 * 48 + tid] = W_f[2 * 1344 + col];
    }
  }
  __syncthreads();

  // each thread owns 2 of the 256 (b,j) outputs; keep its h state in f32 regs
  const int i0 = tid * 2, i1 = tid * 2 + 1;
  float hprev0 = 0.0f, hprev1 = 0.0f;

  for (int t = 0; t < SEQ; ++t) {
    // stage current signals + broadcast h (bf16) into LDS
    if (tid < NB) {
      sigL[tid * 3 + 0] = insig[(tid * SEQ + t) * 2 + 0];
      sigL[tid * 3 + 1] = insig[(tid * SEQ + t) * 2 + 1];
      sigL[tid * 3 + 2] = tcoarse[tid * SEQ + t];
    }
#ifdef HAVE_TDM
    if (wv == 0) {
      tdm_load_2d(lds_hA, h_glob, NB * HID, 1, NB * HID, 1, NB * HID);
      __builtin_amdgcn_s_wait_tensorcnt(0);
    }
#else
    {
      const v4u* s = (const v4u*)h_glob;
      v4u* d = (v4u*)hA;
      for (int r = 0; r < 14; ++r) d[tid + r * 128] = s[tid + r * 128];
    }
#endif
    __syncthreads();

    if (wv < 3) {
      // --- one gate tile per wave: C(16x16) = h(16x896) x W_g(896x16)
      const unsigned short* wbase = wB + wv * 16 * HID;
      v8f c;
      #pragma unroll
      for (int r = 0; r < 8; ++r) c[r] = 0.0f;
      #pragma unroll 4
      for (int ks = 0; ks < HID / 32; ++ks) {
        v16bf a = loadA(hA, lane, ks * 32, HID);
        v16bf b = loadB(wbase, lane, ks * 32, HID);
        c = __builtin_amdgcn_wmma_f32_16x16x32_bf16(false, a, false, b, (short)0, c, false, false);
      }
      int mrow = (lane & 16) ? 8 : 0;
      int ncol = lane & 15;
      #pragma unroll
      for (int r = 0; r < 8; ++r) cT[wv * 256 + (mrow + r) * 16 + ncol] = c[r];
    } else {
      // --- wave 3: gate inputs x = proj(signal) + cond + b_ih  (768 values)
      #pragma unroll 4
      for (int r = 0; r < 24; ++r) {
        int idx = lane + r * 32;         // 0..767
        int g = idx >> 8, rem = idx & 255;
        int b = rem >> 4, jj = rem & 15;
        float v = cond[((size_t)(b * SEQ + t) * 3 + g) * HID + jc + jj] + bihL[g * 16 + jj];
        v += sigL[b * 3 + 0] * wsg[0 * 48 + g * 16 + jj]
           + sigL[b * 3 + 1] * wsg[1 * 48 + g * 16 + jj]
           + sigL[b * 3 + 2] * wsg[2 * 48 + g * 16 + jj];
        xs[idx] = v;
      }
    }
    __syncthreads();

    // --- GRU pointwise combine; write new h slice
    {
      int b0 = i0 >> 4, j0 = i0 & 15;
      float r = fsigm(xs[i0]       + cT[i0]       + bhhL[j0]);
      float z = fsigm(xs[256 + i0] + cT[256 + i0] + bhhL[16 + j0]);
      float n = ftanh(xs[512 + i0] + r * (cT[512 + i0] + bhhL[32 + j0]));
      hprev0 = (1.0f - z) * n + z * hprev0;
      h_glob[b0 * HID + jc + j0] = f2bf(hprev0);
      if (!fine) hc_out[(size_t)(b0 * SEQ + t) * HALF_ + jc + j0] = f2bf(hprev0);
    }
    {
      int b1 = i1 >> 4, j1 = i1 & 15;
      float r = fsigm(xs[i1]       + cT[i1]       + bhhL[j1]);
      float z = fsigm(xs[256 + i1] + cT[256 + i1] + bhhL[16 + j1]);
      float n = ftanh(xs[512 + i1] + r * (cT[512 + i1] + bhhL[32 + j1]));
      hprev1 = (1.0f - z) * n + z * hprev1;
      h_glob[b1 * HID + jc + j1] = f2bf(hprev1);
      if (!fine) hc_out[(size_t)(b1 * SEQ + t) * HALF_ + jc + j1] = f2bf(hprev1);
    }

    // --- device-wide step barrier (monotonic counter, reset by prep kernel)
    __threadfence();
    __syncthreads();
    if (tid == 0) {
      atomicAdd(ctr, 1);
      int target = NWG * (t + 1);
      while (__hip_atomic_load(ctr, __ATOMIC_ACQUIRE, __HIP_MEMORY_SCOPE_AGENT) < target)
        __builtin_amdgcn_s_sleep(1);
    }
    __syncthreads();
  }
}

// ---------------------------------------------------------------- output heads
// out = relu(hc @ W1 + b1) @ W2 + b2 ; one 16-row tile per WG, relu intermediate in LDS
__global__ void heads_kernel(const unsigned short* __restrict__ hc,   // (32768,448) bf16
                             const unsigned short* __restrict__ W1T,  // (448,448) bf16 [n][k]
                             const float* __restrict__ b1,            // (448)
                             const unsigned short* __restrict__ W2T,  // (256,448) bf16 [n][k]
                             const float* __restrict__ b2,            // (256)
                             float* __restrict__ out) {               // (32768,256)
  extern __shared__ unsigned char smem[];
  unsigned short* A  = (unsigned short*)smem;       // [16][448]
  unsigned short* T1 = A + 16 * HALF_;              // [16][448]

  const int m    = blockIdx.x;
  const int tid  = threadIdx.x;
  const int lane = tid & 31;
  const int wv   = tid >> 5;

  // stage the 16-row hidden tile
#ifdef HAVE_TDM
  if (wv == 0) {
    tdm_load_2d((unsigned)(uintptr_t)(void*)A, hc + (size_t)m * 16 * HALF_,
                16 * HALF_, 1, 16 * HALF_, 1, 16 * HALF_);
    __builtin_amdgcn_s_wait_tensorcnt(0);
  }
#else
  {
    const v4u* s = (const v4u*)(hc + (size_t)m * 16 * HALF_);
    v4u* d = (v4u*)A;
    for (int r = 0; r < 7; ++r) d[tid + r * 128] = s[tid + r * 128];
  }
#endif
  __syncthreads();

  // stage 1: T1 = relu(A @ W1 + b1), 28 N-tiles split over 4 waves
  for (int j = 0; j < 7; ++j) {
    int nt = wv * 7 + j;
    v8f c;
    #pragma unroll
    for (int r = 0; r < 8; ++r) c[r] = 0.0f;
    #pragma unroll 2
    for (int ks = 0; ks < HALF_ / 32; ++ks) {
      v16bf a = loadA(A, lane, ks * 32, HALF_);
      v16bf b = loadB(W1T + (size_t)nt * 16 * HALF_, lane, ks * 32, HALF_);
      c = __builtin_amdgcn_wmma_f32_16x16x32_bf16(false, a, false, b, (short)0, c, false, false);
    }
    int mrow = (lane & 16) ? 8 : 0;
    int n = nt * 16 + (lane & 15);
    #pragma unroll
    for (int r = 0; r < 8; ++r) {
      float v = c[r] + b1[n];
      T1[(mrow + r) * HALF_ + n] = f2bf(v > 0.0f ? v : 0.0f);
    }
  }
  __syncthreads();

  // stage 2: out = T1 @ W2 + b2, 16 N-tiles split over 4 waves
  for (int j = 0; j < 4; ++j) {
    int nt = wv * 4 + j;
    v8f c;
    #pragma unroll
    for (int r = 0; r < 8; ++r) c[r] = 0.0f;
    #pragma unroll 2
    for (int ks = 0; ks < HALF_ / 32; ++ks) {
      v16bf a = loadA(T1, lane, ks * 32, HALF_);
      v16bf b = loadB(W2T + (size_t)nt * 16 * HALF_, lane, ks * 32, HALF_);
      c = __builtin_amdgcn_wmma_f32_16x16x32_bf16(false, a, false, b, (short)0, c, false, false);
    }
    int mrow = (lane & 16) ? 8 : 0;
    int n = nt * 16 + (lane & 15);
    #pragma unroll
    for (int r = 0; r < 8; ++r) {
      int row = m * 16 + mrow + r;
      out[(size_t)row * BINS + n] = c[r] + b2[n];
    }
  }
}

// ---------------------------------------------------------------- launch
extern "C" void kernel_launch(void* const* d_in, const int* in_sizes, int n_in,
                              void* d_out, int out_size, void* d_ws, size_t ws_size,
                              hipStream_t stream) {
  const float* cond    = (const float*)d_in[0];
  const float* insig   = (const float*)d_in[1];
  const float* tcoarse = (const float*)d_in[2];
  const float* W_c     = (const float*)d_in[3];
  const float* W_f     = (const float*)d_in[4];
  const float* W_hh    = (const float*)d_in[5];
  const float* b_ih    = (const float*)d_in[6];
  const float* b_hh    = (const float*)d_in[7];
  const float* Wc1     = (const float*)d_in[8];
  const float* bc1     = (const float*)d_in[9];
  const float* Wc2     = (const float*)d_in[10];
  const float* bc2     = (const float*)d_in[11];
  const float* Wf1     = (const float*)d_in[12];
  const float* bf1     = (const float*)d_in[13];
  const float* Wf2     = (const float*)d_in[14];
  const float* bf2     = (const float*)d_in[15];

  char* ws = (char*)d_ws;
  int*            ctr    = (int*)(ws + 0);
  unsigned short* h_glob = (unsigned short*)(ws + 1024);            // 16x896 bf16
  unsigned short* whh_bf = (unsigned short*)(ws + 32768);           // 2688x896 bf16
  unsigned short* Wc1T   = (unsigned short*)(ws + (5ull  << 20));   // 448x448
  unsigned short* Wc2T   = (unsigned short*)(ws + 5767168ull);      // 256x448
  unsigned short* Wf1T   = (unsigned short*)(ws + (6ull  << 20));   // 448x448
  unsigned short* Wf2T   = (unsigned short*)(ws + 6815744ull);      // 256x448
  unsigned short* hc     = (unsigned short*)(ws + (8ull  << 20));   // 32768x448 bf16

  float* out_coarse = (float*)d_out;
  float* out_fine   = out_coarse + (size_t)ROWS * BINS;

  // prep
  zero_init_kernel<<<(NB * HID + 255) / 256, 256, 0, stream>>>(h_glob, ctr);
  conv_bf16_kernel<<<(3 * HID * HID + 255) / 256, 256, 0, stream>>>(W_hh, whh_bf, 3 * HID * HID);
  transpose_bf16_kernel<<<(HALF_ * HALF_ + 255) / 256, 256, 0, stream>>>(Wc1, Wc1T, HALF_, HALF_);
  transpose_bf16_kernel<<<(HALF_ * BINS + 255) / 256, 256, 0, stream>>>(Wc2, Wc2T, HALF_, BINS);
  transpose_bf16_kernel<<<(HALF_ * HALF_ + 255) / 256, 256, 0, stream>>>(Wf1, Wf1T, HALF_, HALF_);
  transpose_bf16_kernel<<<(HALF_ * BINS + 255) / 256, 256, 0, stream>>>(Wf2, Wf2T, HALF_, BINS);

  // GRU scan: 56 persistent WGs, 122 KB LDS each (weights resident)
  const int gru_lds = 121984;
  (void)hipFuncSetAttribute((const void*)gru_kernel,
                            hipFuncAttributeMaxDynamicSharedMemorySize, gru_lds);
  gru_kernel<<<NWG, 128, gru_lds, stream>>>(cond, insig, tcoarse, W_c, W_f, b_ih, b_hh,
                                            whh_bf, h_glob, hc, ctr);

  // output heads
  const int head_lds = 2 * 16 * HALF_ * 2; // 28672
  heads_kernel<<<ROWS / 16, 128, head_lds, stream>>>(hc, Wc1T, bc1, Wc2T, bc2, out_coarse);
  heads_kernel<<<ROWS / 16, 128, head_lds, stream>>>(hc, Wf1T, bf1, Wf2T, bf2, out_fine);
}
